// QCrossPatch_73349451481437
// MI455X (gfx1250) — 8-bit matrix operand order repaired and checked by reading, compiled, not run on hardware
//
#include <hip/hip_runtime.h>
#include <math.h>

typedef __attribute__((ext_vector_type(8))) int   v8i;
typedef __attribute__((ext_vector_type(4))) int   v4i;
typedef __attribute__((ext_vector_type(4))) float v4f;

#define NB    256
#define NP    196
#define ND    384
#define PPAD  208
#define KPAD  256
#define MT    13
#define NT    24
#define KT    4
#define XFRAG_BYTES (KT * NT * 1024)
#define AFRAG_BYTES (MT * KT * 1024)
#define SMEM_BYTES  (XFRAG_BYTES + AFRAG_BYTES)

__device__ __forceinline__ int a_off(int mtile, int kc, int m, int kin) {
  int v    = ((kin >> 4) << 1) | ((kin >> 2) & 1);
  int lane = (m & 15) | (((kin >> 3) & 1) << 4);
  return (mtile * KT + kc) * 1024 + lane * 32 + v * 4 + (kin & 3);
}

__global__ void prep_w2_kernel(const float* __restrict__ attn_w,
                               const float* __restrict__ w2_scale,
                               const float* __restrict__ act_scale,
                               signed char* __restrict__ wsA,
                               float* __restrict__ out_tail) {
  int idx = blockIdx.x * 256 + threadIdx.x;
  if (idx == 0) out_tail[0] = w2_scale[0] * act_scale[0];
  if (idx >= PPAD * KPAD) return;
  int p = idx / KPAD, q = idx % KPAD;
  float v = 0.0f;
  if (p < NP && q < NP) {
    float w = attn_w[p * NP + q] / w2_scale[0];
    w = fminf(fmaxf(w, -128.0f), 127.0f);
    v = rintf(w);
  }
  wsA[a_off(p >> 4, q >> 6, p & 15, q & 63)] = (signed char)(int)v;
}

__device__ __forceinline__ unsigned q8(float v, float a_s, float s) {
  float r = rintf((v / a_s) / s);
  r = fminf(fmaxf(r, -128.0f), 127.0f);
  return (unsigned)(unsigned char)(signed char)(int)r;
}

__global__ void __launch_bounds__(256, 1)
qgemm_kernel(const float* __restrict__ x,
             const float* __restrict__ a_s_p,
             const float* __restrict__ act_scale_p,
             const float* __restrict__ w2_scale_p,
             const signed char* __restrict__ wsA,
             float* __restrict__ out) {
  extern __shared__ char smem[];
  char* ldsX = smem;
  char* ldsA = smem + XFRAG_BYTES;
  const int tid = threadIdx.x;
  const int b   = blockIdx.x;

  const float a_s  = a_s_p[0];
  const float ascl = act_scale_p[0];
  const float b2_s = w2_scale_p[0] * ascl;
  float ratio = ascl / a_s;
  float inv   = 1.0f / ratio;
  int e; float m = frexpf(inv, &e);
  float mult = rintf(m * 65536.0f);
  float s = exp2f((float)(16 - e)) / mult;

  for (int i = tid; i < XFRAG_BYTES / 16; i += 256) {
    v4i z = {0, 0, 0, 0};
    ((v4i*)ldsX)[i] = z;
  }
  {
    unsigned ldsA_base = (unsigned)(unsigned long long)(ldsA);
    for (int j = tid; j < AFRAG_BYTES / 16; j += 256) {
      unsigned loff = ldsA_base + j * 16;
      unsigned long long g = (unsigned long long)(wsA + j * 16);
      asm volatile("global_load_async_to_lds_b128 %0, %1, off"
                   :: "v"(loff), "v"(g) : "memory");
    }
  }
  __syncthreads();

  const float* xb = x + (size_t)b * (NP * ND);
  for (int j = tid; j < (NP / 4) * (ND / 4); j += 256) {
    int q4 = j / 96;
    int d  = (j - q4 * 96) * 4;
    int q  = q4 * 4;
    v4f r0 = *(const v4f*)(xb + (q + 0) * ND + d);
    v4f r1 = *(const v4f*)(xb + (q + 1) * ND + d);
    v4f r2 = *(const v4f*)(xb + (q + 2) * ND + d);
    v4f r3 = *(const v4f*)(xb + (q + 3) * ND + d);
    int kc  = q >> 6;
    int kin = q & 63;
    int v     = ((kin >> 4) << 1) | ((kin >> 2) & 1);
    int lane0 = (d & 15) | (((kin >> 3) & 1) << 4);
    int base  = ((kc * NT + (d >> 4)) * 1024) + lane0 * 32 + v * 4;
#pragma unroll
    for (int c = 0; c < 4; ++c) {
      unsigned w = q8(r0[c], a_s, s)
                 | (q8(r1[c], a_s, s) << 8)
                 | (q8(r2[c], a_s, s) << 16)
                 | (q8(r3[c], a_s, s) << 24);
      *(unsigned*)(ldsX + base + c * 32) = w;
    }
  }
  asm volatile("s_wait_asynccnt 0" ::: "memory");
  __syncthreads();

  const int wave = tid >> 5;
  const int lane = tid & 31;
  float* ob = out + (size_t)b * (NP * ND);

  auto store_tile = [&](const v8i& acc, int mtile, int ntile) {
    int n = ntile * 16 + (lane & 15);
    if (mtile < 12) {
      int mbase = mtile * 16 + ((lane >> 4) << 3);
#pragma unroll
      for (int r = 0; r < 8; ++r)
        ob[(size_t)(mbase + r) * ND + n] = (float)acc[r] * b2_s;
    } else if (lane < 16) {
#pragma unroll
      for (int r = 0; r < 4; ++r)
        ob[(size_t)(192 + r) * ND + n] = (float)acc[r] * b2_s;
    }
  };

  for (int mtile = 0; mtile < MT; ++mtile) {
    v8i acc0 = {0, 0, 0, 0, 0, 0, 0, 0};
    v8i acc1 = {0, 0, 0, 0, 0, 0, 0, 0};
    v8i acc2 = {0, 0, 0, 0, 0, 0, 0, 0};
#pragma unroll
    for (int kc = 0; kc < KT; ++kc) {
      v8i afr = *(const v8i*)(ldsA + (mtile * KT + kc) * 1024 + lane * 32);
      v8i bf0 = *(const v8i*)(ldsX + (kc * NT + wave) * 1024 + lane * 32);
      v8i bf1 = *(const v8i*)(ldsX + (kc * NT + wave + 8) * 1024 + lane * 32);
      v8i bf2 = *(const v8i*)(ldsX + (kc * NT + wave + 16) * 1024 + lane * 32);
      acc0 = __builtin_amdgcn_wmma_i32_16x16x64_iu8(true, afr, true, bf0, acc0, false, false);
      acc1 = __builtin_amdgcn_wmma_i32_16x16x64_iu8(true, afr, true, bf1, acc1, false, false);
      acc2 = __builtin_amdgcn_wmma_i32_16x16x64_iu8(true, afr, true, bf2, acc2, false, false);
    }
    store_tile(acc0, mtile, wave);
    store_tile(acc1, mtile, wave + 8);
    store_tile(acc2, mtile, wave + 16);
  }
}

extern "C" void kernel_launch(void* const* d_in, const int* in_sizes, int n_in,
                              void* d_out, int out_size, void* d_ws, size_t ws_size,
                              hipStream_t stream) {
  const float* x         = (const float*)d_in[0];
  const float* a_s       = (const float*)d_in[1];
  const float* attn_w    = (const float*)d_in[4];
  const float* w2_scale  = (const float*)d_in[8];
  const float* act_scale = (const float*)d_in[9];
  signed char* wsA = (signed char*)d_ws;
  float* out = (float*)d_out;

  prep_w2_kernel<<<dim3((PPAD * KPAD + 255) / 256), dim3(256), 0, stream>>>(
      attn_w, w2_scale, act_scale, wsA, out + (size_t)NB * NP * ND);

  qgemm_kernel<<<dim3(NB), dim3(256), SMEM_BYTES, stream>>>(
      x, a_s, act_scale, w2_scale, wsA, out);
}
